// VanillaAttention_60421599920395
// MI455X (gfx1250) — matrix operand order repaired and checked by reading, not compiled, not run on hardware
//
#include <hip/hip_runtime.h>
#include <cstdint>

// ---------------------------------------------------------------------------
// Causal MHA block for MI455X (gfx1250, wave32, WMMA + TDM).
// x(B,T,C) @ w_attn + b -> qkv ; flash attention per head ; y @ w_proj + b.
// GEMM math on v_wmma_f32_16x16x32_f16; A-tiles DMA'd into LDS by the
// Tensor Data Mover (TENSORcnt), double-buffered against WMMA compute.
// Attention: Bc=64 key blocks, log2-domain online softmax.
// ---------------------------------------------------------------------------

typedef _Float16 f16;
typedef __attribute__((ext_vector_type(16))) _Float16 v16h;
typedef __attribute__((ext_vector_type(8)))  float    v8f;
typedef __attribute__((ext_vector_type(4)))  unsigned int u32x4;
typedef __attribute__((ext_vector_type(8)))  int i32x8;
typedef __attribute__((ext_vector_type(4)))  int i32x4;

#if defined(__HIP_DEVICE_COMPILE__) && defined(__has_builtin)
# if __has_builtin(__builtin_amdgcn_tensor_load_to_lds)
#  define HAVE_TDM 1
# endif
#endif
#ifndef HAVE_TDM
# define HAVE_TDM 0
#endif

union Frag {            // one WMMA 16-bit A/B operand: 16 halves = 8 VGPRs
  v16h  v;
  uint4 q[2];
};

__device__ __forceinline__ v8f v8f_zero() {
  v8f z = {0.f,0.f,0.f,0.f,0.f,0.f,0.f,0.f};
  return z;
}

__device__ __forceinline__ v8f wmma16(const Frag& a, const Frag& b, v8f c) {
  return __builtin_amdgcn_wmma_f32_16x16x32_f16(false, a.v, false, b.v,
                                                (short)0, c, false, false);
}

#if HAVE_TDM
// 2D TDM tile load: tile_x elements (data_size=2B) per row, tile_y rows,
// row stride = stride_elems; LDS padding per D# group1 pad codes.
__device__ __forceinline__ void tdm_load_2d(uint32_t lds_byte_addr,
                                            const void* gaddr,
                                            uint32_t tile_x, uint32_t tile_y,
                                            uint32_t stride_elems,
                                            uint32_t tensor_d0, uint32_t tensor_d1,
                                            uint32_t pad_int_code, uint32_t pad_amt_code)
{
  uint64_t ga = (uint64_t)(uintptr_t)gaddr;
  u32x4 g0;
  g0.x = 1u;                                   // count=1 (user mode)
  g0.y = lds_byte_addr;                        // lds_addr
  g0.z = (uint32_t)ga;                         // global_addr[31:0]
  g0.w = (uint32_t)(ga >> 32) | (2u << 30);    // global_addr[56:32] | type=2
  i32x8 g1;
  g1[0] = (int)((1u << 16)                     // data_size = 2B
              | (1u << 20)                     // pad_enable
              | (pad_int_code << 22)
              | (pad_amt_code << 25));
  g1[1] = (int)((tensor_d0 & 0xFFFFu) << 16);                  // dim0[15:0]
  g1[2] = (int)((tensor_d0 >> 16) | ((tensor_d1 & 0xFFFFu) << 16));
  g1[3] = (int)((tensor_d1 >> 16) | (tile_x << 16));           // tile_dim0
  g1[4] = (int)(tile_y & 0xFFFFu);                             // tile_dim1
  g1[5] = (int)stride_elems;                                   // dim0_stride[31:0]
  g1[6] = 0;
  g1[7] = 0;
  i32x4 z4 = {0,0,0,0};
#if __clang_major__ >= 23
  i32x8 z8 = {0,0,0,0,0,0,0,0};
  __builtin_amdgcn_tensor_load_to_lds(g0, g1, z4, z4, z8, 0);
#else
  __builtin_amdgcn_tensor_load_to_lds(g0, g1, z4, z4, 0);
#endif
}
#endif

// ---------------------------------------------------------------------------
// f32 -> f16 conversion
// ---------------------------------------------------------------------------
__global__ void cvt_f32_f16_k(const float* __restrict__ in,
                              f16* __restrict__ out, int n) {
  int i = blockIdx.x * blockDim.x + threadIdx.x;
  if (i < n) out[i] = (f16)in[i];
}

// ---------------------------------------------------------------------------
// Tiled f16 GEMM:  C[M,N] = A[M,K] @ B[K,N] + bias[N]
// 128x128x32 block tile, 256 threads = 8 waves, wave tile = 64x32 (8 WMMA acc).
// A tiles: TDM DMA into double-buffered LDS (row pitch 40 halves via TDM pad).
// B tiles: double-buffered manual transpose into Bs[n][k].
// ---------------------------------------------------------------------------
template<bool OUT_F32>
__global__ __launch_bounds__(256)
void gemm_f16_k(const f16* __restrict__ A, const f16* __restrict__ Bm,
                const float* __restrict__ bias, void* __restrict__ Cout,
                int M, int N, int K)
{
  constexpr int BM = 128, BN = 128, BK = 32;
  constexpr int LDA = BK + 8;        // 40 halves = 80B pitch (TDM pad 4 DW / 16 DW)
  constexpr int LDB = BK + 8;
  constexpr int ABUF = BM * LDA;
  constexpr int BBUF = BN * LDB;
  extern __shared__ f16 smem[];      // 40960 bytes
  f16* As = smem;
  f16* Bs = smem + 2 * ABUF;

  const int tid   = threadIdx.x;
  const int lane  = tid & 31;
  const int wave  = tid >> 5;
  const int row   = lane & 15;
  const int hi    = lane >> 4;
  const int waveM = (wave >> 2) * 64;
  const int waveN = (wave & 3) * 32;
  const int bM = blockIdx.y * BM;
  const int bN = blockIdx.x * BN;
  const uint32_t ldsBase = __builtin_amdgcn_groupstaticsize();

  auto stageB = [&](int buf, int k0) {
    f16* bs = Bs + buf * BBUF;
    #pragma unroll
    for (int t = 0; t < 2; ++t) {
      int idx = tid + t * 256;
      int r = idx >> 4, c = (idx & 15) * 8;
      uint4 d = *(const uint4*)(Bm + (size_t)(k0 + r) * N + bN + c);
      f16 tmp[8]; *(uint4*)tmp = d;
      #pragma unroll
      for (int e = 0; e < 8; ++e) bs[(c + e) * LDB + r] = tmp[e];
    }
  };

  auto issueA = [&](int buf, int k0) {
#if HAVE_TDM
    if (wave == 0) {
      tdm_load_2d(ldsBase + (uint32_t)(buf * ABUF * 2),
                  A + (size_t)bM * K + k0,
                  (uint32_t)BK, (uint32_t)BM, (uint32_t)K,
                  (uint32_t)K, (uint32_t)M, 3u, 3u);
    }
#else
    #pragma unroll
    for (int t = 0; t < 2; ++t) {
      int idx = tid + t * 256;
      int r = idx >> 2, c = (idx & 3) * 8;
      uint32_t dst = ldsBase + (uint32_t)(buf * ABUF * 2) + (uint32_t)((r * LDA + c) * 2);
      unsigned long long src =
          (unsigned long long)(uintptr_t)(A + (size_t)(bM + r) * K + k0 + c);
      asm volatile("global_load_async_to_lds_b128 %0, %1, off"
                   :: "v"(dst), "v"(src) : "memory");
    }
#endif
  };

  auto waitA = [&]() {
#if HAVE_TDM
    if (wave == 0) __builtin_amdgcn_s_wait_tensorcnt(0);
#else
    asm volatile("s_wait_asynccnt 0x0" ::: "memory");
#endif
  };

  v8f acc[4][2];
  #pragma unroll
  for (int i = 0; i < 4; ++i)
    #pragma unroll
    for (int j = 0; j < 2; ++j)
      acc[i][j] = v8f_zero();

  const int NK = K / BK;

  issueA(0, 0);
  stageB(0, 0);
  waitA();
  __syncthreads();

  for (int kt = 0; kt < NK; ++kt) {
    const int cur = kt & 1, nxt = cur ^ 1;
    if (kt + 1 < NK) {
      issueA(nxt, (kt + 1) * BK);
      stageB(nxt, (kt + 1) * BK);
      __builtin_prefetch(Bm + (size_t)((kt + 1) * BK + (tid >> 3)) * N + bN + (tid & 7) * 16, 0, 1);
    }

    const f16* as = As + cur * ABUF;
    const f16* bs = Bs + cur * BBUF;
    Frag af[4], bf[2];
    #pragma unroll
    for (int mf = 0; mf < 4; ++mf) {
      const f16* ap = as + (waveM + mf * 16 + row) * LDA + hi * 8;
      af[mf].q[0] = *(const uint4*)ap;
      af[mf].q[1] = *(const uint4*)(ap + 16);
    }
    #pragma unroll
    for (int nf = 0; nf < 2; ++nf) {
      const f16* bp = bs + (waveN + nf * 16 + row) * LDB + hi * 8;
      bf[nf].q[0] = *(const uint4*)bp;
      bf[nf].q[1] = *(const uint4*)(bp + 16);
    }
    #pragma unroll
    for (int mf = 0; mf < 4; ++mf)
      #pragma unroll
      for (int nf = 0; nf < 2; ++nf)
        acc[mf][nf] = wmma16(af[mf], bf[nf], acc[mf][nf]);

    if (kt + 1 < NK) waitA();
    __syncthreads();
  }

  #pragma unroll
  for (int mf = 0; mf < 4; ++mf)
    #pragma unroll
    for (int nf = 0; nf < 2; ++nf) {
      int n = bN + waveN + nf * 16 + row;
      float bv = bias[n];
      #pragma unroll
      for (int i = 0; i < 8; ++i) {
        int m = bM + waveM + mf * 16 + i + 8 * hi;
        float val = acc[mf][nf][i] + bv;
        if (OUT_F32) ((float*)Cout)[(size_t)m * N + n] = val;
        else         ((f16*)Cout)[(size_t)m * N + n]   = (f16)val;
      }
    }
}

// ---------------------------------------------------------------------------
// Flash attention, causal. One wave = 16 query rows; block = 8 waves = 128
// query rows; Bc = 64 keys per block (amortizes softmax reductions, barriers
// and V staging over 2x keys vs Bc=32; WMMA:VALU ratio improves ~1.5x).
// Per key-block: S = Q(16x64) @ K^T -> 8 WMMA; log2-domain online softmax;
// P via per-wave LDS (C-layout -> A-layout + f16 cvt); O += P @ V -> 8 WMMA.
// ---------------------------------------------------------------------------
__global__ __launch_bounds__(256)
void attn_flash_k(const f16* __restrict__ qkv, f16* __restrict__ y)
{
  constexpr int T = 2048, D = 64, C = 1024, C3 = 3072;
  constexpr int BC = 64;
  constexpr int LDP = 72, LDV = 72;  // 144B pitch: keeps b128 reads aligned
  __shared__ f16 Vt[D * LDV];        // V tile transposed: [dim][key]
  __shared__ f16 Pt[8 * 16 * LDP];   // per-wave P staging [wave][qrow][key]

  const int tid  = threadIdx.x;
  const int lane = tid & 31;
  const int wave = tid >> 5;
  const int row  = lane & 15;
  const int hi   = lane >> 4;

  int bid = blockIdx.x;
  const int qblk = bid & 15;  bid >>= 4;   // T/128 = 16
  const int h    = bid & 15;  bid >>= 4;   // H = 16
  const int b    = bid;
  const int q0 = qblk * 128;
  const int qw = q0 + wave * 16;

  const size_t bOff = (size_t)b * T * C3;
  const f16* qp    = qkv + bOff + (size_t)(qw + row) * C3 + h * D;
  const f16* kBase = qkv + bOff + C     + h * D;
  const f16* vBase = qkv + bOff + 2 * C + h * D;

  Frag qf[2];
  #pragma unroll
  for (int kc = 0; kc < 2; ++kc) {
    qf[kc].q[0] = *(const uint4*)(qp + kc * 32 + hi * 8);
    qf[kc].q[1] = *(const uint4*)(qp + kc * 32 + hi * 8 + 16);
  }

  float mrow[8], lrow[8];
  v8f o[4];
  #pragma unroll
  for (int i = 0; i < 8; ++i) { mrow[i] = -3.0e38f; lrow[i] = 0.f; }
  #pragma unroll
  for (int nd = 0; nd < 4; ++nd) o[nd] = v8f_zero();

  const int nkb = q0 / BC + 2;          // 64-key blocks covering q0..q0+127
  for (int jb = 0; jb < nkb; ++jb) {
    const int kb = jb * BC;

    {   // stage V tile 64x64, transposed into Vt[dim][key]
      int key = tid >> 2;               // 0..63
      int d0  = (tid & 3) * 16;         // 0,16,32,48
      const f16* vp = vBase + (size_t)(kb + key) * C3 + d0;
      uint4 dv0 = *(const uint4*)vp;
      uint4 dv1 = *(const uint4*)(vp + 8);
      f16 tmp[16];
      *(uint4*)tmp = dv0; *(uint4*)(tmp + 8) = dv1;
      #pragma unroll
      for (int e = 0; e < 16; ++e) Vt[(d0 + e) * LDV + key] = tmp[e];
    }
    __syncthreads();

    if (kb <= qw + 15) {
      // ---- S = Q(16x64) @ K^T(64x64) : 8 WMMA ----
      v8f s[4];
      #pragma unroll
      for (int nb = 0; nb < 4; ++nb) {
        s[nb] = v8f_zero();
        const f16* kp = kBase + (size_t)(kb + nb * 16 + row) * C3;
        #pragma unroll
        for (int kc = 0; kc < 2; ++kc) {
          Frag kf;
          kf.q[0] = *(const uint4*)(kp + kc * 32 + hi * 8);
          kf.q[1] = *(const uint4*)(kp + kc * 32 + hi * 8 + 16);
          s[nb] = wmma16(qf[kc], kf, s[nb]);
        }
      }

      // ---- log2-domain online softmax ----
      const float scale2 = 0.1803368801111244f;   // log2(e)/sqrt(64)
      #pragma unroll
      for (int i = 0; i < 8; ++i) {
        const int qg = qw + i + 8 * hi;
        float a[4];
        #pragma unroll
        for (int nb = 0; nb < 4; ++nb) {
          a[nb] = s[nb][i] * scale2;
          if (kb + nb * 16 + row > qg) a[nb] = -3.0e38f;
        }
        float mx = fmaxf(fmaxf(a[0], a[1]), fmaxf(a[2], a[3]));
        mx = fmaxf(mx, __shfl_xor(mx, 1, 32));
        mx = fmaxf(mx, __shfl_xor(mx, 2, 32));
        mx = fmaxf(mx, __shfl_xor(mx, 4, 32));
        mx = fmaxf(mx, __shfl_xor(mx, 8, 32));
        const float mnew = fmaxf(mrow[i], mx);
        const float corr = exp2f(mrow[i] - mnew);
        float p[4], sum = 0.f;
        #pragma unroll
        for (int nb = 0; nb < 4; ++nb) { p[nb] = exp2f(a[nb] - mnew); sum += p[nb]; }
        sum += __shfl_xor(sum, 1, 32);
        sum += __shfl_xor(sum, 2, 32);
        sum += __shfl_xor(sum, 4, 32);
        sum += __shfl_xor(sum, 8, 32);
        lrow[i] = lrow[i] * corr + sum;
        mrow[i] = mnew;
        #pragma unroll
        for (int nb = 0; nb < 4; ++nb) s[nb][i] = p[nb];
        #pragma unroll
        for (int nd = 0; nd < 4; ++nd) o[nd][i] *= corr;
      }

      // ---- P: C layout -> A layout via per-wave LDS tile (f16 cvt) ----
      f16* pw = &Pt[wave * 16 * LDP];
      #pragma unroll
      for (int nb = 0; nb < 4; ++nb)
        #pragma unroll
        for (int i = 0; i < 8; ++i)
          pw[(i + 8 * hi) * LDP + nb * 16 + row] = (f16)s[nb][i];
      __builtin_amdgcn_wave_barrier();

      Frag pf[2];
      #pragma unroll
      for (int kp = 0; kp < 2; ++kp) {
        const f16* pr = pw + row * LDP + kp * 32 + hi * 8;
        pf[kp].q[0] = *(const uint4*)pr;
        pf[kp].q[1] = *(const uint4*)(pr + 16);
      }

      // ---- O += P(16x64) @ V(64x64) : 8 WMMA ----
      #pragma unroll
      for (int nd = 0; nd < 4; ++nd) {
        #pragma unroll
        for (int kv = 0; kv < 2; ++kv) {
          Frag vf;
          const f16* vp = &Vt[(nd * 16 + row) * LDV + kv * 32 + hi * 8];
          vf.q[0] = *(const uint4*)vp;
          vf.q[1] = *(const uint4*)(vp + 16);
          o[nd] = wmma16(pf[kv], vf, o[nd]);
        }
      }
    }
    __syncthreads();
  }

  #pragma unroll
  for (int nd = 0; nd < 4; ++nd)
    #pragma unroll
    for (int i = 0; i < 8; ++i) {
      const int qg = qw + i + 8 * hi;
      float val = o[nd][i] / lrow[i];
      y[((size_t)b * T + qg) * C + h * D + nd * 16 + row] = (f16)val;
    }
}

// ---------------------------------------------------------------------------
// Host launcher
// ---------------------------------------------------------------------------
extern "C" void kernel_launch(void* const* d_in, const int* in_sizes, int n_in,
                              void* d_out, int out_size, void* d_ws, size_t ws_size,
                              hipStream_t stream) {
  (void)in_sizes; (void)n_in; (void)out_size; (void)ws_size;
  const float* x      = (const float*)d_in[0];
  const float* w_attn = (const float*)d_in[1];
  const float* b_attn = (const float*)d_in[2];
  const float* w_proj = (const float*)d_in[3];
  const float* b_proj = (const float*)d_in[4];
  float* out = (float*)d_out;

  const int Bv = 2, T = 2048, C = 1024;
  const int M = Bv * T;            // 4096
  const size_t GEMM_LDS = 2 * 128 * 40 * sizeof(f16) * 2;  // 40960 bytes

  char* ws = (char*)d_ws;
  f16* xh   = (f16*)(ws);                                 // 8 MiB
  f16* wah  = (f16*)(ws + (size_t)8  * 1024 * 1024);      // 6 MiB
  f16* wph  = (f16*)(ws + (size_t)14 * 1024 * 1024);      // 2 MiB
  f16* qkvh = (f16*)(ws + (size_t)16 * 1024 * 1024);      // 24 MiB
  f16* yh   = (f16*)(ws + (size_t)40 * 1024 * 1024);      // 8 MiB

  cvt_f32_f16_k<<<(M * C + 255) / 256, 256, 0, stream>>>(x, xh, M * C);
  cvt_f32_f16_k<<<(C * 3 * C + 255) / 256, 256, 0, stream>>>(w_attn, wah, C * 3 * C);
  cvt_f32_f16_k<<<(C * C + 255) / 256, 256, 0, stream>>>(w_proj, wph, C * C);

  gemm_f16_k<false><<<dim3(3 * C / 128, M / 128), 256, GEMM_LDS, stream>>>(
      xh, wah, b_attn, (void*)qkvh, M, 3 * C, C);

  attn_flash_k<<<dim3(Bv * 16 * (T / 128)), 256, 0, stream>>>(qkvh, yh);

  gemm_f16_k<true><<<dim3(C / 128, M / 128), 256, GEMM_LDS, stream>>>(
      yh, wph, b_proj, (void*)out, M, C, C);
}
